// Attention_21595095564821
// MI455X (gfx1250) — compile-verified, matrix operand order checked by reading, not run on hardware
//
#include <hip/hip_runtime.h>

typedef __attribute__((ext_vector_type(16))) __bf16 v16bf;
typedef __attribute__((ext_vector_type(8)))  float  v8f;
typedef __attribute__((ext_vector_type(8)))  __bf16 v8bf;
typedef __attribute__((ext_vector_type(4)))  __bf16 v4bf;

#define SEQ  4096
#define FEAT 1024
#define N3   3072
#define NH   16
#define HD   64
#define LOG2_10000 13.287712379549449f

// ---------------------------------------------------------------------------
// CDNA5 async global->LDS copy (16B per lane), tracked by ASYNCcnt.
// VFLAT syntax: global_load_async_to_lds_b128 vdst(LDS byte offset), vaddr64, off
// LDS byte offset = low 32 bits of the generic LDS pointer (flat->LDS rule).
// Async loads complete IN ORDER (ISA 7.1) -> partial waits enable
// double-buffered copy/compute overlap.
// ---------------------------------------------------------------------------
static __device__ __forceinline__ void async_copy16(void* lds, const void* g) {
    asm volatile("global_load_async_to_lds_b128 %0, %1, off"
                 :
                 : "v"((unsigned int)(uintptr_t)lds),
                   "v"((unsigned long long)(uintptr_t)g)
                 : "memory");
}
#define ASYNC_WAIT(n) asm volatile("s_wait_asynccnt %0" :: "i"(n) : "memory")

// ---------------------------------------------------------------------------
// WMMA fragment helper (CDNA5 wave32, 16x16x32 bf16; layout per ISA 7.12.2)
// A (16x32, MxK): lane holds row r = lane&15; VGPR v holds K pair
//   k0 = (v>=4 ? 16:0) + (lane>=16 ? 8:0) + (v&3)*2
// Per lane: row[hi8..hi8+7] and row[16+hi8..16+hi8+7] -> 2x 16B contiguous
// chunks -> ds_load_b128 x2. All operands use this (B via transposed storage).
// C/D (16x16 f32): lane holds col c = lane&15; VGPR v holds row v + 8*(lane>=16)
// ---------------------------------------------------------------------------
static __device__ __forceinline__ v16bf load_frag_A(const __bf16* p, int ld, int lane) {
    const int r   = lane & 15;
    const int hi8 = (lane >> 4) << 3;
    const __bf16* row = p + r * ld;
    v16bf f;
#pragma unroll
    for (int v = 0; v < 8; ++v) {
        int k0 = ((v & 4) << 2) + hi8 + ((v & 3) << 1);
        f[2 * v]     = row[k0];
        f[2 * v + 1] = row[k0 + 1];
    }
    return f;
}

static __device__ __forceinline__ v8f wmma_bf16(v16bf a, v16bf b, v8f c) {
    return __builtin_amdgcn_wmma_f32_16x16x32_bf16(
        /*neg_a=*/false, a, /*neg_b=*/false, b,
        /*c_mod=*/(short)0, c, /*reuse_a=*/false, /*reuse_b=*/false);
}

// ---------------------------------------------------------------------------
// Kernel 1a: fp32 -> bf16 conversion (vectorized x4), same layout
// ---------------------------------------------------------------------------
__global__ __launch_bounds__(256) void cvt_bf16_kernel(const float* __restrict__ src,
                                                       v4bf* __restrict__ dst, int n4) {
    int i = blockIdx.x * 256 + threadIdx.x;
    if (i >= n4) return;
    float4 v = reinterpret_cast<const float4*>(src)[i];
    v4bf o;
    o[0] = (__bf16)v.x; o[1] = (__bf16)v.y; o[2] = (__bf16)v.z; o[3] = (__bf16)v.w;
    dst[i] = o;
}

// ---------------------------------------------------------------------------
// Kernel 1b: fp32 [K][N] -> bf16 transposed [N][K]  (W -> W^T, one-time)
// ---------------------------------------------------------------------------
__global__ __launch_bounds__(256) void cvt_bf16_transpose_kernel(
    const float* __restrict__ src, __bf16* __restrict__ dst, int K, int N) {
    int i = blockIdx.x * 256 + threadIdx.x;          // one float4 along N
    int n4 = N >> 2;
    if (i >= K * n4) return;
    int k = i / n4, n = (i - k * n4) << 2;
    float4 v = reinterpret_cast<const float4*>(src + (size_t)k * N)[n >> 2];
    dst[(size_t)(n + 0) * K + k] = (__bf16)v.x;
    dst[(size_t)(n + 1) * K + k] = (__bf16)v.y;
    dst[(size_t)(n + 2) * K + k] = (__bf16)v.z;
    dst[(size_t)(n + 3) * K + k] = (__bf16)v.w;
}

// ---------------------------------------------------------------------------
// Kernel 2: qkv = x @ W + b, RoPE fused into epilogue for q/k.
// BM=128, BN=64, BK=32; 256 threads = 8 waves. Double-buffered async staging:
// next K-slab flies while current slab's WMMAs run (wait asynccnt<=3).
// q pre-scaled by 1/sqrt(64) (exact in bf16). q,k -> [head][seq][64];
// v -> TRANSPOSED [head][feat][seq] with packed 16B stores.
// ---------------------------------------------------------------------------
__global__ __launch_bounds__(256) void qkv_rope_kernel(
    const __bf16* __restrict__ xb, const __bf16* __restrict__ wt,
    const float* __restrict__ bias,
    __bf16* __restrict__ qb, __bf16* __restrict__ kb, __bf16* __restrict__ vt) {

    __shared__ __align__(16) __bf16 As[2][128 * 32];  // 16 KB, [m][k]
    __shared__ __align__(16) __bf16 Bt[2][64 * 32];   //  8 KB, [n][k]

    const int tid  = threadIdx.x;
    const int lane = tid & 31;
    const int w    = tid >> 5;
    const int n0   = blockIdx.x * 64;
    const int m0   = blockIdx.y * 128;

    // issue 3 async instructions per wave: 2 for A slab, 1 for B slab
    auto stage = [&](int k0, __bf16* Ad, __bf16* Bd) {
#pragma unroll
        for (int i = 0; i < 2; ++i) {
            int c = tid + i * 256, row = c >> 2, cc = c & 3;
            async_copy16(reinterpret_cast<uint4*>(Ad) + c,
                         reinterpret_cast<const uint4*>(
                             xb + (size_t)(m0 + row) * FEAT + k0) + cc);
        }
        async_copy16(reinterpret_cast<uint4*>(Bd) + tid,
                     reinterpret_cast<const uint4*>(
                         wt + (size_t)(n0 + (tid >> 2)) * FEAT + k0) + (tid & 3));
    };

    v8f acc[4] = {};
    const int NK = FEAT / 32;  // 32 K-slabs

    stage(0, As[0], Bt[0]);
    for (int it = 0; it < NK; ++it) {
        const int cur = it & 1;
        const int knext = ((it + 1 < NK) ? (it + 1) : it) * 32;  // clamped re-stage
        __syncthreads();                       // prev compute done reading buf[1-cur]
        stage(knext, As[1 - cur], Bt[1 - cur]);
        {
            int kpf = ((it + 2 < NK) ? (it + 2) : it) * 32;      // prefetch slab+2
            __builtin_prefetch(xb + (size_t)(m0 + (tid >> 1)) * FEAT + kpf, 0, 3);
            __builtin_prefetch(wt + (size_t)(n0 + (tid & 63)) * FEAT + kpf, 0, 3);
        }
        ASYNC_WAIT(3);                         // current slab's 3 copies landed
        __syncthreads();

        v16bf a = load_frag_A(As[cur] + (w * 16) * 32, 32, lane);
#pragma unroll
        for (int t = 0; t < 4; ++t) {
            v16bf b = load_frag_A(Bt[cur] + (t * 16) * 32, 32, lane);
            acc[t] = wmma_bf16(a, b, acc[t]);
        }
    }

    // ---- epilogue: bias + (RoPE for q,k; q *= 0.125) + bf16 store ----
    const int c   = lane & 15;
    const int hi8 = (lane >> 4) << 3;
    const int qkv  = n0 >> 10;          // 0=q 1=k 2=v
    const int head = (n0 & 1023) >> 6;

#pragma unroll
    for (int t = 0; t < 4; ++t) {
        float bv = bias[n0 + t * 16 + c];
#pragma unroll
        for (int v = 0; v < 8; ++v) acc[t][v] += bv;
    }

    const size_t hbase = (size_t)head * SEQ * HD;
    if (qkv < 2) {
        // RoPE: pair (feature j, j+32) == (tile t2, tile t2+2) same lane/slot
        float its0 = exp2f(-(float)(c)      * (LOG2_10000 / 32.0f));
        float its1 = exp2f(-(float)(c + 16) * (LOG2_10000 / 32.0f));
#pragma unroll
        for (int v = 0; v < 8; ++v) {
            float p = (float)(m0 + w * 16 + v + hi8);
            {
                float ang = p * its0, s = __sinf(ang), co = __cosf(ang);
                float f1 = acc[0][v], f2 = acc[2][v];
                acc[0][v] = f1 * co - f2 * s;
                acc[2][v] = f2 * co + f1 * s;
            }
            {
                float ang = p * its1, s = __sinf(ang), co = __cosf(ang);
                float f1 = acc[1][v], f2 = acc[3][v];
                acc[1][v] = f1 * co - f2 * s;
                acc[3][v] = f2 * co + f1 * s;
            }
        }
        if (qkv == 0) {  // fold softmax scale 1/sqrt(64) into q (exact in bf16)
#pragma unroll
            for (int t = 0; t < 4; ++t)
#pragma unroll
                for (int v = 0; v < 8; ++v) acc[t][v] *= 0.125f;
        }
        __bf16* dst = (qkv == 0) ? qb : kb;
#pragma unroll
        for (int t = 0; t < 4; ++t) {
            int f = t * 16 + c;
#pragma unroll
            for (int v = 0; v < 8; ++v) {
                int row = m0 + w * 16 + v + hi8;
                dst[hbase + (size_t)row * HD + f] = (__bf16)acc[t][v];
            }
        }
    } else {
        // v: transposed [head][feat][seq]; 8 consecutive seq elems per lane
        const int row0 = m0 + w * 16 + hi8;
#pragma unroll
        for (int t = 0; t < 4; ++t) {
            int f = t * 16 + c;
            v8bf o;
#pragma unroll
            for (int v = 0; v < 8; ++v) o[v] = (__bf16)acc[t][v];
            *reinterpret_cast<v8bf*>(vt + hbase + (size_t)f * SEQ + row0) = o;
        }
    }
}

// ---------------------------------------------------------------------------
// Kernel 3: causal flash attention. grid = (SEQ/64, NH), 128 threads = 4 waves,
// wave w owns 16 query rows. Double-buffered async staging of K / V^T tiles
// (wait asynccnt<=8); causal mask only on the diagonal tile (scores arrive
// pre-scaled via q); online softmax; P round-trips per-wave LDS (same-wave
// DS ordering, no barrier needed).
// ---------------------------------------------------------------------------
__global__ __launch_bounds__(128) void flash_attn_kernel(
    const __bf16* __restrict__ qb, const __bf16* __restrict__ kb,
    const __bf16* __restrict__ vt, float* __restrict__ out) {

    __shared__ __align__(16) __bf16 Ks[2][64 * 64];   // 16 KB, [key][feat]
    __shared__ __align__(16) __bf16 Vt[2][64 * 64];   // 16 KB, [feat][key]
    __shared__ __align__(16) __bf16 Ps[4][16 * 64];   //  8 KB, per-wave [row][key]

    const int tid  = threadIdx.x;
    const int lane = tid & 31;
    const int w    = tid >> 5;
    const int c    = lane & 15;
    const int hi8  = (lane >> 4) << 3;
    const int q0   = blockIdx.x * 64;
    const int h    = blockIdx.y;

    const __bf16* qh = qb + (size_t)h * SEQ * HD;
    const __bf16* kh = kb + (size_t)h * SEQ * HD;
    const __bf16* vh = vt + (size_t)h * SEQ * HD;   // [feat][seq]

    // issue 8 async instructions per wave: 4 K chunks + 4 V^T chunks
    auto stage = [&](int kb2, __bf16* Kd, __bf16* Vd) {
#pragma unroll
        for (int i = 0; i < 4; ++i) {
            int ch = tid + i * 128, row = ch >> 3, cc = ch & 7;
            async_copy16(reinterpret_cast<uint4*>(Kd) + ch,
                         reinterpret_cast<const uint4*>(
                             kh + (size_t)(kb2 + row) * HD) + cc);
            async_copy16(reinterpret_cast<uint4*>(Vd) + ch,
                         reinterpret_cast<const uint4*>(
                             vh + (size_t)row * SEQ + kb2) + cc);
        }
    };

    // Q fragments for this wave's 16 rows (K-dim 64 -> 2 frags), kept in VGPRs
    const __bf16* qrowp = qh + (size_t)(q0 + w * 16) * HD;
    v16bf qf0 = load_frag_A(qrowp, HD, lane);
    v16bf qf1 = load_frag_A(qrowp + 32, HD, lane);

    v8f O[4] = {};
    float mrow[8], lrow[8];
#pragma unroll
    for (int v = 0; v < 8; ++v) { mrow[v] = -3.0e38f; lrow[v] = 0.0f; }

    const int nkt = (q0 >> 6) + 1;  // causal: key tiles 0 .. q0/64
    stage(0, Ks[0], Vt[0]);
    for (int kt = 0; kt < nkt; ++kt) {
        const int cur   = kt & 1;
        const int kbase = kt * 64;
        const int knext = (kt + 1 < nkt) ? (kbase + 64) : kbase;  // clamped
        __syncthreads();                 // prev compute done reading buf[1-cur]
        stage(knext, Ks[1 - cur], Vt[1 - cur]);
        {
            int kpf = (kt + 2 < nkt) ? (kbase + 128) : kbase;     // prefetch tile+2
            __builtin_prefetch(kh + (size_t)(kpf + (tid & 63)) * HD, 0, 3);
            __builtin_prefetch(vh + (size_t)(tid & 63) * SEQ + kpf, 0, 3);
        }
        ASYNC_WAIT(8);                   // current tile's 8 copies landed
        __syncthreads();

        const __bf16* Kc = Ks[cur];
        const __bf16* Vc = Vt[cur];

        // S = Q K^T (pre-scaled by 1/8 via q); K^T frag = A-style load
        v8f S[4] = {};
#pragma unroll
        for (int t = 0; t < 4; ++t) {
            v16bf b0 = load_frag_A(Kc + (t * 16) * HD, HD, lane);
            v16bf b1 = load_frag_A(Kc + (t * 16) * HD + 32, HD, lane);
            S[t] = wmma_bf16(qf0, b0, S[t]);
            S[t] = wmma_bf16(qf1, b1, S[t]);
        }

        // causal mask: only the diagonal tile needs it
        if (kt == nkt - 1) {
#pragma unroll
            for (int t = 0; t < 4; ++t) {
                int key = kbase + t * 16 + c;
#pragma unroll
                for (int v = 0; v < 8; ++v) {
                    int row = q0 + w * 16 + v + hi8;
                    S[t][v] = (key <= row) ? S[t][v] : -3.0e38f;
                }
            }
        }

        // online softmax per row (rows live in 16-lane half-groups)
#pragma unroll
        for (int v = 0; v < 8; ++v) {
            float m = fmaxf(fmaxf(S[0][v], S[1][v]), fmaxf(S[2][v], S[3][v]));
#pragma unroll
            for (int msk = 1; msk <= 8; msk <<= 1) m = fmaxf(m, __shfl_xor(m, msk, 32));
            float mnew = fmaxf(mrow[v], m);
            float corr = __expf(mrow[v] - mnew);
            float rs = 0.0f;
#pragma unroll
            for (int t = 0; t < 4; ++t) {
                float p = __expf(S[t][v] - mnew);
                S[t][v] = p;
                rs += p;
            }
#pragma unroll
            for (int msk = 1; msk <= 8; msk <<= 1) rs += __shfl_xor(rs, msk, 32);
            lrow[v] = lrow[v] * corr + rs;
            mrow[v] = mnew;
#pragma unroll
            for (int t = 0; t < 4; ++t) O[t][v] *= corr;
        }

        // P (C-layout) -> per-wave LDS -> A-layout frags (same-wave DS order)
        __bf16* Pw = Ps[w];
#pragma unroll
        for (int t = 0; t < 4; ++t)
#pragma unroll
            for (int v = 0; v < 8; ++v)
                Pw[(v + hi8) * 64 + t * 16 + c] = (__bf16)S[t][v];

        v16bf pf0 = load_frag_A(Pw, 64, lane);
        v16bf pf1 = load_frag_A(Pw + 32, 64, lane);
#pragma unroll
        for (int t = 0; t < 4; ++t) {
            // B[key][feat] = Vt[feat][key] -> A-style loads on Vt
            v16bf b0 = load_frag_A(Vc + (t * 16) * HD, HD, lane);       // keys 0..31
            v16bf b1 = load_frag_A(Vc + (t * 16) * HD + 32, HD, lane);  // keys 32..63
            O[t] = wmma_bf16(pf0, b0, O[t]);
            O[t] = wmma_bf16(pf1, b1, O[t]);
        }
    }

    // normalize (rcp + mul) and store: out[row][h*64+f] (lanes 0..15 contiguous)
    float rinv[8];
#pragma unroll
    for (int v = 0; v < 8; ++v) rinv[v] = __builtin_amdgcn_rcpf(lrow[v]);
#pragma unroll
    for (int t = 0; t < 4; ++t) {
        int f = h * HD + t * 16 + c;
#pragma unroll
        for (int v = 0; v < 8; ++v) {
            int row = q0 + w * 16 + v + hi8;
            out[(size_t)row * FEAT + f] = O[t][v] * rinv[v];
        }
    }
}

// ---------------------------------------------------------------------------
extern "C" void kernel_launch(void* const* d_in, const int* in_sizes, int n_in,
                              void* d_out, int out_size, void* d_ws, size_t ws_size,
                              hipStream_t stream) {
    const float* x    = (const float*)d_in[0];  // (4096, 1024)
    const float* Wkqv = (const float*)d_in[1];  // (1024, 3072)
    const float* bkqv = (const float*)d_in[2];  // (3072,)
    float* out = (float*)d_out;                 // (4096, 1024)

    char* ws = (char*)d_ws;
    const size_t QB = (size_t)NH * SEQ * HD * sizeof(__bf16);   // 8 MB each
    __bf16* qb = (__bf16*)(ws);
    __bf16* kb = (__bf16*)(ws + QB);
    __bf16* vt = (__bf16*)(ws + 2 * QB);                        // [head][feat][seq]
    __bf16* xb = (__bf16*)(ws + 3 * QB);                        // 8 MB
    __bf16* wt = (__bf16*)(ws + 3 * QB + (size_t)SEQ * FEAT * sizeof(__bf16)); // 6 MB, W^T

    // 1) fp32 -> bf16 conversions (W transposed)
    {
        int n4 = SEQ * FEAT / 4;
        cvt_bf16_kernel<<<(n4 + 255) / 256, 256, 0, stream>>>(x, (v4bf*)xb, n4);
    }
    {
        int total4 = FEAT * N3 / 4;
        cvt_bf16_transpose_kernel<<<(total4 + 255) / 256, 256, 0, stream>>>(
            Wkqv, wt, FEAT, N3);
    }

    // 2) qkv GEMM + fused RoPE (+ transposed V store)
    {
        dim3 grid(N3 / 64, SEQ / 128);
        qkv_rope_kernel<<<grid, 256, 0, stream>>>(xb, wt, bkqv, qb, kb, vt);
    }

    // 3) causal flash attention
    {
        dim3 grid(SEQ / 64, NH);
        flash_attn_kernel<<<grid, 128, 0, stream>>>(qb, kb, vt, out);
    }
}
